// Encoder_5222680232495
// MI455X (gfx1250) — matrix operand order repaired and checked by reading, not compiled, not run on hardware
//
#include <hip/hip_runtime.h>
#include <hip/hip_bf16.h>
#include <math.h>

typedef __attribute__((ext_vector_type(16))) _Float16 v16h;
typedef __attribute__((ext_vector_type(8)))  _Float16 v8h;
typedef __attribute__((ext_vector_type(8)))  float    v8f;
typedef __attribute__((ext_vector_type(4)))  float    v4f;
typedef __attribute__((ext_vector_type(8)))  float    v8fx;
typedef __attribute__((ext_vector_type(4)))  unsigned int v4u;
typedef __attribute__((ext_vector_type(8)))  int      v8i;
typedef __attribute__((ext_vector_type(4)))  int      v4i;

#define DD 128            // feature dim
#define LL 3
#define GG 512
#define WT_STRIDE 136     // padded row stride (halves) for transposed W
#define WT_ELEMS  (DD * WT_STRIDE)   // 17408 halves = 34816 bytes per matrix

// ---------------------------------------------------------------- utilities
__global__ void zero_kernel(float* __restrict__ p, int n) {
    int i = blockIdx.x * blockDim.x + threadIdx.x;
    if (i < n) p[i] = 0.0f;
}

__global__ void copy_kernel(float* __restrict__ dst, const float* __restrict__ src, int n) {
    int i = blockIdx.x * blockDim.x + threadIdx.x;
    if (i < n) dst[i] = src[i];
}

// Pre-transpose + convert all 6 weight matrices to f16, padded stride.
// wt layout: [6][DD][WT_STRIDE] halves; mats 0..2 = W1[l], 3..5 = W2[l].
__global__ void transpose_weights_f16_kernel(const float* __restrict__ W1,
                                             const float* __restrict__ W2,
                                             _Float16* __restrict__ wt) {
    int tid = blockIdx.x * blockDim.x + threadIdx.x;
    if (tid >= 6 * DD * DD) return;
    int mat = tid / (DD * DD);
    int idx = tid % (DD * DD);
    int k = idx >> 7;
    int n = idx & 127;
    const float* src = (mat < 3) ? (W1 + (size_t)mat * DD * DD)
                                 : (W2 + (size_t)(mat - 3) * DD * DD);
    wt[(size_t)mat * WT_ELEMS + n * WT_STRIDE + k] = (_Float16)src[idx];
}

// ------------------------------------------------- edge scatter-add (GIN agg)
// agg must be pre-initialized with h (self term, eps=0).
// One thread per (edge, 4-feature chunk): coalesced float4 gather + 4 f32 atomics.
__global__ void scatter_add_kernel(const float* __restrict__ h,
                                   const long long* __restrict__ src,
                                   const long long* __restrict__ dst,
                                   float* __restrict__ agg, int E_) {
    int tid = blockIdx.x * blockDim.x + threadIdx.x;
    int e  = tid >> 5;          // 32 threads per edge (128 feats / 4)
    int c4 = (tid & 31) << 2;
    if (e >= E_) return;
    int s = (int)src[e];
    int d = (int)dst[e];
    const float4 v = *(const float4*)(h + (size_t)s * DD + c4);
    float* p = agg + (size_t)d * DD + c4;
    atomicAdd(p + 0, v.x);
    atomicAdd(p + 1, v.y);
    atomicAdd(p + 2, v.z);
    atomicAdd(p + 3, v.w);
}

// --------------------------------------------------- WMMA GEMM + bias + ReLU
// Out[N,128] = relu(A[N,128] @ W[128,128] + bias).
// Wt is pre-transposed f16 [n][k] (padded stride) in global; each block DMAs
// its 34 KB W tile into LDS via the Tensor Data Mover (TENSORcnt), then
// 8 waves each compute a 16x128 strip with v_wmma_f32_16x16x32_f16.
__global__ __launch_bounds__(256) void gemm_bias_relu_wmma(
    const float* __restrict__ A, const _Float16* __restrict__ Wt,
    const float* __restrict__ bias, float* __restrict__ Out, int Nrows)
{
    __shared__ __align__(16) _Float16 Wlds[WT_ELEMS];  // 34 KB, Wt[n][k]
    const int tid     = threadIdx.x;
    const int rowBase = blockIdx.x * DD;
    const int wave    = tid >> 5;

#if __has_builtin(__builtin_amdgcn_tensor_load_to_lds)
    // ---- TDM: one 1-D descriptor DMA of the whole W tile, issued by wave 0.
    if (wave == 0) {
        unsigned long long ga = (unsigned long long)(uintptr_t)Wt;
        // D# group 0: count=1 | lds_addr=0 (Wlds is the only LDS object) |
        //             global_addr | type=2 ("image")
        v4u g0 = { 1u,                                  // count=1, user mode
                   0u,                                  // lds_addr (bytes)
                   (unsigned int)(ga & 0xFFFFFFFFu),
                   (unsigned int)((ga >> 32) & 0x01FFFFFFu) | (2u << 30) };
        // D# group 1: data_size=2B, 1-D tensor/tile of WT_ELEMS elements
        const unsigned int td0 = (unsigned int)WT_ELEMS;   // 17408
        v8i g1 = { (int)0x00010000u,                       // data_size=1 (2B)
                   (int)((td0 & 0xFFFFu) << 16),           // tensor_dim0 lo16
                   (int)(((td0 >> 16) & 0xFFFFu) | (1u << 16)), // dim0 hi | tensor_dim1=1
                   (int)((td0 & 0xFFFFu) << 16),           // tile_dim0 = td0
                   0,                                      // tile_dim1/2 unused
                   (int)td0,                               // tensor_dim0_stride lo32
                   0, 0 };                                 // stride hi, dim1_stride
        v4i g2 = { 0, 0, 0, 0 };
        v4i g3 = { 0, 0, 0, 0 };
#if defined(__clang_major__) && __clang_major__ >= 23
        v8i g4 = { 0, 0, 0, 0, 0, 0, 0, 0 };
        __builtin_amdgcn_tensor_load_to_lds(g0, g1, g2, g3, g4, 0);
#else
        __builtin_amdgcn_tensor_load_to_lds(g0, g1, g2, g3, 0);
#endif
        __builtin_amdgcn_s_wait_tensorcnt(0);
    }
    // Anchor store: the TDM intrinsic's LDS write isn't tied to Wlds, so give
    // the compiler a real store (into the never-read pad region k=128..135)
    // to prevent "load of unwritten shared global" -> undef folding.
    if (tid == 0) {
        *((volatile _Float16*)(Wlds + 128)) = (_Float16)0.0f;
    }
#else
    // Fallback: plain vectorized cooperative copy (Wt already f16/transposed)
    for (int idx = tid; idx < WT_ELEMS / 8; idx += 256)
        ((v8h*)Wlds)[idx] = ((const v8h*)Wt)[idx];
#endif
    __syncthreads();
    asm volatile("" ::: "memory");

    const int lane = tid & 31;
    const int m16  = lane & 15;   // M (A row) / N (B col) within 16x16 tile
    const int grp  = lane >> 4;   // half-wave group
    const int rloc = wave * 16;

    // A-fragment source row for this lane; clamp to a valid row (no branches),
    // out-of-range results are suppressed at the store side.
    const int gra   = rowBase + rloc + m16;
    const int graC  = (gra < Nrows) ? gra : (Nrows - 1);
    const float* __restrict__ arow = A + (size_t)graC * DD;

    v8f acc[8] = {};              // 8 col-tiles of 16x16 f32

#pragma unroll
    for (int kc = 0; kc < 4; ++kc) {      // K = 128 in chunks of 32
        const int kb = kc * 32;

        // ---- A fragment: two contiguous 8-float runs of this lane's row
        // halves [0..7]  = K kb + grp*8 .. +7      (VGPRs 0..3)
        // halves [8..15] = K kb + 16 + grp*8 .. +7 (VGPRs 4..7)
        const float* p0 = arow + kb + grp * 8;
        v4f f0 = *(const v4f*)(p0);
        v4f f1 = *(const v4f*)(p0 + 4);
        v4f f2 = *(const v4f*)(p0 + 16);
        v4f f3 = *(const v4f*)(p0 + 20);

        v8fx flo = __builtin_shufflevector(f0, f1, 0, 1, 2, 3, 4, 5, 6, 7);
        v8fx fhi = __builtin_shufflevector(f2, f3, 0, 1, 2, 3, 4, 5, 6, 7);
        v8h  alo = __builtin_convertvector(flo, v8h);
        v8h  ahi = __builtin_convertvector(fhi, v8h);
        v16h a   = __builtin_shufflevector(alo, ahi,
                     0, 1, 2, 3, 4, 5, 6, 7, 8, 9, 10, 11, 12, 13, 14, 15);

#pragma unroll
        for (int t = 0; t < 8; ++t) {     // 8 column tiles
            // ---- B fragment: Wt[col][kb+grp*8 ..+7] and Wt[col][kb+16+grp*8 ..+7]
            const v8h* bp = (const v8h*)&Wlds[(t * 16 + m16) * WT_STRIDE
                                              + kb + grp * 8];
            v8h  b0 = bp[0];
            v8h  b1 = bp[2];
            v16h b  = __builtin_shufflevector(b0, b1,
                        0, 1, 2, 3, 4, 5, 6, 7, 8, 9, 10, 11, 12, 13, 14, 15);
            acc[t] = __builtin_amdgcn_wmma_f32_16x16x32_f16(
                false, a, false, b, (short)0, acc[t], false, false);
        }
    }

    // epilogue: bias + ReLU; C/D layout: row = grp*8 + v, col = t*16 + m16
#pragma unroll
    for (int t = 0; t < 8; ++t) {
        const int col = t * 16 + m16;
        const float bv = bias[col];
#pragma unroll
        for (int v = 0; v < 8; ++v) {
            int gr = rowBase + rloc + grp * 8 + v;
            if (gr < Nrows) {
                float val = acc[t][v] + bv;
                Out[(size_t)gr * DD + col] = val > 0.0f ? val : 0.0f;
            }
        }
    }
}

// ------------------------------------------------------------- BN statistics
__global__ void bn_stats_kernel(const float* __restrict__ z,
                                float* __restrict__ sums,
                                float* __restrict__ sumsq, int Nrows) {
    const int col  = threadIdx.x;
    const int row0 = blockIdx.x * 256;
    const int row1 = (row0 + 256 < Nrows) ? row0 + 256 : Nrows;
    float s = 0.0f, ss = 0.0f;
    for (int r = row0; r < row1; ++r) {
        float v = z[(size_t)r * DD + col];
        s  += v;
        ss += v * v;
    }
    atomicAdd(&sums[col], s);
    atomicAdd(&sumsq[col], ss);
}

// ------------------------------------- BN apply + graph pooling (+ final h)
__global__ void bn_apply_pool_kernel(const float* __restrict__ z,
                                     const float* __restrict__ sums,
                                     const float* __restrict__ sumsq,
                                     const float* __restrict__ gamma,
                                     const float* __restrict__ beta,
                                     const long long* __restrict__ batch,
                                     float* __restrict__ hout,
                                     float* __restrict__ pooled,
                                     float* __restrict__ hfinal,
                                     int Nrows, int l) {
    int tid = blockIdx.x * blockDim.x + threadIdx.x;
    if (tid >= Nrows * DD) return;
    int row = tid >> 7;
    int col = tid & 127;
    float invN = 1.0f / (float)Nrows;
    float mu   = sums[col] * invN;
    float var  = sumsq[col] * invN - mu * mu;
    float val  = gamma[col] * (z[tid] - mu) * rsqrtf(var + 1e-5f) + beta[col];
    hout[tid] = val;
    if (hfinal) hfinal[tid] = val;
    int g = (int)batch[row];
    atomicAdd(&pooled[(size_t)g * (LL * DD) + l * DD + col], val);
}

// ----------------------------------------------------------------- launcher
extern "C" void kernel_launch(void* const* d_in, const int* in_sizes, int n_in,
                              void* d_out, int out_size, void* d_ws, size_t ws_size,
                              hipStream_t stream) {
    const float*     x     = (const float*)d_in[0];
    const long long* eidx  = (const long long*)d_in[1];   // int64 in reference
    const long long* batch = (const long long*)d_in[2];
    const float*     W1    = (const float*)d_in[3];
    const float*     b1    = (const float*)d_in[4];
    const float*     W2    = (const float*)d_in[5];
    const float*     b2    = (const float*)d_in[6];
    const float*     gamma = (const float*)d_in[7];
    const float*     beta  = (const float*)d_in[8];

    const int N = in_sizes[0] / DD;
    const int E = in_sizes[1] / 2;
    const long long* src = eidx;
    const long long* dst = eidx + E;

    const size_t nd = (size_t)N * DD;
    float*     hbuf   = (float*)d_ws;
    float*     aggbuf = hbuf + nd;
    float*     z1     = aggbuf + nd;
    float*     z2     = z1 + nd;
    float*     sums   = z2 + nd;            // [128]
    float*     sumsq  = sums + DD;          // [128]
    _Float16*  wt     = (_Float16*)(sumsq + DD);  // [6][DD][WT_STRIDE] f16

    float* pooled = (float*)d_out;                 // [G, L*D]
    float* hfinal = pooled + (size_t)GG * LL * DD; // [N, D]

    // zero the pooled output region (harness poisons d_out)
    const int pn = GG * LL * DD;
    zero_kernel<<<(pn + 255) / 256, 256, 0, stream>>>(pooled, pn);

    // pre-transpose/convert all weights once
    const int wn = 6 * DD * DD;
    transpose_weights_f16_kernel<<<(wn + 255) / 256, 256, 0, stream>>>(W1, W2, wt);

    const int ndI     = (int)nd;
    const int gblocks = (N + DD - 1) / DD;
    const int sthr    = E * 32;

    for (int l = 0; l < LL; ++l) {
        const float* hcur = (l == 0) ? x : hbuf;

        // agg = h (self term), then agg += sum of neighbor features
        copy_kernel<<<(ndI + 255) / 256, 256, 0, stream>>>(aggbuf, hcur, ndI);
        scatter_add_kernel<<<(sthr + 255) / 256, 256, 0, stream>>>(hcur, src, dst, aggbuf, E);

        // MLP: relu(agg@W1+b1) -> relu(.@W2+b2)   (WMMA f16->f32, TDM W tiles)
        gemm_bias_relu_wmma<<<gblocks, 256, 0, stream>>>(
            aggbuf, wt + (size_t)l * WT_ELEMS, b1 + l * DD, z1, N);
        gemm_bias_relu_wmma<<<gblocks, 256, 0, stream>>>(
            z1, wt + (size_t)(3 + l) * WT_ELEMS, b2 + l * DD, z2, N);

        // BatchNorm (training-mode batch stats) + pooled segment-sum
        zero_kernel<<<1, 256, 0, stream>>>(sums, 2 * DD);  // sums & sumsq contiguous
        bn_stats_kernel<<<(N + 255) / 256, DD, 0, stream>>>(z2, sums, sumsq, N);

        const int bt = N * DD;
        bn_apply_pool_kernel<<<(bt + 255) / 256, 256, 0, stream>>>(
            z2, sums, sumsq, gamma + l * DD, beta + l * DD, batch,
            hbuf, pooled, (l == LL - 1) ? hfinal : nullptr, N, l);
    }
}
